// GAT_CL_61658550502129
// MI455X (gfx1250) — matrix operand order repaired and checked by reading, not compiled, not run on hardware
//
#include <hip/hip_runtime.h>
#include <hip/hip_bf16.h>

// GAT (DGI-style, 4 streams x 2 layers) for MI455X / gfx1250.
// Dense h = x@W uses v_wmma_f32_16x16x32_bf16 (bf16 in, f32 acc).
// Edge softmax + aggregation use hardware f32 atomics; h (25.6MB) is L2-resident.

#define GAT_N 50000
#define GAT_D 128
#define GAT_E 600000
#define GAT_L 2

typedef __attribute__((ext_vector_type(16))) __bf16 v16bf;
typedef __attribute__((ext_vector_type(8)))  __bf16 v8bf;
typedef __attribute__((ext_vector_type(8)))  float  v8f;

// ---------------------------------------------------------------------------
// GEMM: h[N,128] = x[N,128] @ W[128,128]   (bf16 WMMA, f32 accumulate)
// Block = 256 threads = 8 waves; each wave computes a 16x128 strip of h.
// W is transposed into LDS as bf16 once per block.
// ---------------------------------------------------------------------------
__global__ __launch_bounds__(256) void gat_gemm(const float* __restrict__ x,
                                                const float* __restrict__ W,
                                                float* __restrict__ h,
                                                int nRows) {
    __shared__ alignas(16) __bf16 Wt[128][136];  // padded: 272B rows -> bank spread

    const int tid = threadIdx.x;
    // Transpose W (fp32 row-major [k][n]) -> LDS bf16 Wt[n][k]
    for (int idx = tid; idx < 128 * 128; idx += 256) {
        const int k = idx >> 7;
        const int n = idx & 127;
        Wt[n][k] = (__bf16)W[idx];
    }
    __syncthreads();

    const int wave  = tid >> 5;
    const int lane  = tid & 31;
    const int half  = lane >> 4;   // 0 or 1
    const int l16   = lane & 15;
    const int strip = blockIdx.x * 128 + wave * 16;
    const int arow  = strip + l16;                // row this lane feeds into A
    const bool rowOK = (arow < nRows);
    const float* xrow = x + (long)arow * GAT_D;

    v8f c[8];
#pragma unroll
    for (int nt = 0; nt < 8; ++nt)
#pragma unroll
        for (int j = 0; j < 8; ++j) c[nt][j] = 0.0f;

#pragma unroll
    for (int kt = 0; kt < 4; ++kt) {
        // ---- A fragment: 16x32 bf16.
        // lanes 0-15 (half=0): K = kt*32 + [0..7]   and kt*32 + [16..23]
        // lanes16-31 (half=1): K = kt*32 + [8..15]  and kt*32 + [24..31]
        const int ka = kt * 32 + half * 8;
        v16bf a;
        if (rowOK) {
            const float4 f0 = *(const float4*)(xrow + ka);
            const float4 f1 = *(const float4*)(xrow + ka + 4);
            const float4 f2 = *(const float4*)(xrow + ka + 16);
            const float4 f3 = *(const float4*)(xrow + ka + 20);
            a[0]  = (__bf16)f0.x; a[1]  = (__bf16)f0.y; a[2]  = (__bf16)f0.z; a[3]  = (__bf16)f0.w;
            a[4]  = (__bf16)f1.x; a[5]  = (__bf16)f1.y; a[6]  = (__bf16)f1.z; a[7]  = (__bf16)f1.w;
            a[8]  = (__bf16)f2.x; a[9]  = (__bf16)f2.y; a[10] = (__bf16)f2.z; a[11] = (__bf16)f2.w;
            a[12] = (__bf16)f3.x; a[13] = (__bf16)f3.y; a[14] = (__bf16)f3.z; a[15] = (__bf16)f3.w;
        } else {
#pragma unroll
            for (int i = 0; i < 16; ++i) a[i] = (__bf16)0.0f;
        }

        // ---- B fragments: 32x16 bf16 per output tile.
        // lane = n + 16*half holds column n, K = kt*32 + 8*half + [0..7] and kt*32 + 8*half + [16..23]
        const int kb = kt * 32 + half * 8;
#pragma unroll
        for (int nt = 0; nt < 8; ++nt) {
            const int col = nt * 16 + l16;
            const v8bf blo = *(const v8bf*)&Wt[col][kb];
            const v8bf bhi = *(const v8bf*)&Wt[col][kb + 16];
            v16bf b;
#pragma unroll
            for (int i = 0; i < 8; ++i) { b[i] = blo[i]; b[i + 8] = bhi[i]; }
            c[nt] = __builtin_amdgcn_wmma_f32_16x16x32_bf16(
                false, a, false, b, (short)0, c[nt], false, false);
        }
    }

    // ---- C/D layout: lane group (half) -> rows M = half*8 + r; N = l16 per tile
#pragma unroll
    for (int nt = 0; nt < 8; ++nt) {
        const int col = nt * 16 + l16;
#pragma unroll
        for (int r = 0; r < 8; ++r) {
            const int row = strip + half * 8 + r;
            if (row < nRows) h[(long)row * GAT_D + col] = c[nt][r];
        }
    }
}

// ---------------------------------------------------------------------------
// Per-node attention scalars: alpha_s[n] = h[n].a_src ; alpha_d[n] = h[n].a_dst
// One wave per node, float4 per lane, shuffle reduction.
// ---------------------------------------------------------------------------
__global__ __launch_bounds__(256) void node_alpha(const float* __restrict__ h,
                                                  const float* __restrict__ avs,
                                                  const float* __restrict__ avd,
                                                  float* __restrict__ as_out,
                                                  float* __restrict__ ad_out,
                                                  int n) {
    const int wave = threadIdx.x >> 5;
    const int lane = threadIdx.x & 31;
    const int node = blockIdx.x * 8 + wave;
    if (node >= n) return;
    const float4 hv = *(const float4*)(h + (long)node * GAT_D + lane * 4);
    const float4 a0 = *(const float4*)(avs + lane * 4);
    const float4 d0 = *(const float4*)(avd + lane * 4);
    float ps = hv.x * a0.x + hv.y * a0.y + hv.z * a0.z + hv.w * a0.w;
    float pd = hv.x * d0.x + hv.y * d0.y + hv.z * d0.z + hv.w * d0.w;
#pragma unroll
    for (int m = 16; m > 0; m >>= 1) {
        ps += __shfl_xor(ps, m, 32);
        pd += __shfl_xor(pd, m, 32);
    }
    if (lane == 0) { as_out[node] = ps; ad_out[node] = pd; }
}

// ---------------------------------------------------------------------------
// Edge logits + segment max (order-preserving u32-encoded float atomicMax)
// ---------------------------------------------------------------------------
__device__ __forceinline__ unsigned enc_f32(float f) {
    unsigned u = __float_as_uint(f);
    return (u & 0x80000000u) ? ~u : (u | 0x80000000u);
}
__device__ __forceinline__ float dec_f32(unsigned u) {
    u = (u & 0x80000000u) ? (u & 0x7FFFFFFFu) : ~u;
    return __uint_as_float(u);
}

__global__ void edge_logits(const int* __restrict__ esrc, const int* __restrict__ edst,
                            const float* __restrict__ as_in, const float* __restrict__ ad_in,
                            float* __restrict__ logit, unsigned* __restrict__ mEnc,
                            int Etot, int Ereal) {
    const int t = blockIdx.x * blockDim.x + threadIdx.x;
    if (t >= Etot) return;
    const int sN = (t < Ereal) ? esrc[t] : (t - Ereal);  // self loops appended
    const int dN = (t < Ereal) ? edst[t] : (t - Ereal);
    float v = as_in[sN] + ad_in[dN];
    v = (v >= 0.0f) ? v : 0.2f * v;   // leaky_relu, slope 0.2
    logit[t] = v;
    atomicMax(&mEnc[dN], enc_f32(v));
}

// ---------------------------------------------------------------------------
// e = exp(logit - m[dst]) (in place) + segment sum via f32 atomics
// ---------------------------------------------------------------------------
__global__ void edge_exp(const int* __restrict__ edst, float* __restrict__ le,
                         const unsigned* __restrict__ mEnc, float* __restrict__ sSum,
                         int Etot, int Ereal) {
    const int t = blockIdx.x * blockDim.x + threadIdx.x;
    if (t >= Etot) return;
    const int dN = (t < Ereal) ? edst[t] : (t - Ereal);
    const float m  = dec_f32(mEnc[dN]);
    const float ee = __expf(le[t] - m);
    le[t] = ee;
    atomicAdd(&sSum[dN], ee);
}

// ---------------------------------------------------------------------------
// Aggregation: acc[dst] += h[src] * (e / s[dst]).  One wave per edge:
// 32 lanes x float4 = full 128-float row; scatter via global_atomic_add_f32.
// h/acc are L2-resident (25.6MB each << 192MB L2).
// ---------------------------------------------------------------------------
__global__ __launch_bounds__(256) void edge_aggregate(const int* __restrict__ esrc,
                                                      const int* __restrict__ edst,
                                                      const float* __restrict__ h,
                                                      const float* __restrict__ e,
                                                      const float* __restrict__ sSum,
                                                      float* __restrict__ acc,
                                                      int Etot, int Ereal) {
    const int wid  = (blockIdx.x * blockDim.x + threadIdx.x) >> 5;
    const int lane = threadIdx.x & 31;
    if (wid >= Etot) return;
    const int sN = (wid < Ereal) ? esrc[wid] : (wid - Ereal);
    const int dN = (wid < Ereal) ? edst[wid] : (wid - Ereal);
    const float coef = e[wid] / sSum[dN];
    const float4 hv = *(const float4*)(h + (long)sN * GAT_D + lane * 4);
    float* dst = acc + (long)dN * GAT_D + lane * 4;
    atomicAdd(dst + 0, hv.x * coef);
    atomicAdd(dst + 1, hv.y * coef);
    atomicAdd(dst + 2, hv.z * coef);
    atomicAdd(dst + 3, hv.w * coef);
}

// ---------------------------------------------------------------------------
// out = prelu(acc + b, w)   (elementwise; safe in-place)
// ---------------------------------------------------------------------------
__global__ void gat_finalize(const float* __restrict__ acc, const float* __restrict__ b,
                             const float* __restrict__ pw, float* __restrict__ out,
                             int total) {
    const int t = blockIdx.x * blockDim.x + threadIdx.x;
    if (t >= total) return;
    const float v = acc[t] + b[t & (GAT_D - 1)];
    const float w = pw[0];
    out[t] = (v >= 0.0f) ? v : w * v;
}

__global__ void zero_u32(unsigned* __restrict__ p, long n) {
    const long t = (long)blockIdx.x * blockDim.x + threadIdx.x;
    if (t < n) p[t] = 0u;
}

// ---------------------------------------------------------------------------
// Orchestration: 4 streams x 2 layers, all on `stream`, graph-capture safe.
// Workspace (~55MB): bufA (layer I/O + acc), hbuf, edge buf, node scalars.
// ---------------------------------------------------------------------------
extern "C" void kernel_launch(void* const* d_in, const int* in_sizes, int n_in,
                              void* d_out, int out_size, void* d_ws, size_t ws_size,
                              hipStream_t stream) {
    (void)n_in; (void)out_size; (void)ws_size;
    const int N = in_sizes[0] / GAT_D;        // 50000
    const int E = in_sizes[1] / 2;            // 600000
    const int Etot = E + N;

    const float* X = (const float*)d_in[0];
    const int* edges[4] = { (const int*)d_in[1], (const int*)d_in[2],
                            (const int*)d_in[3], (const int*)d_in[4] };
    const float* Wset[2]  = { (const float*)d_in[5],  (const float*)d_in[9]  };
    const float* ASset[2] = { (const float*)d_in[6],  (const float*)d_in[10] };
    const float* ADset[2] = { (const float*)d_in[7],  (const float*)d_in[11] };
    const float* Bset[2]  = { (const float*)d_in[8],  (const float*)d_in[12] };
    const float* pw = (const float*)d_in[13];
    float* out = (float*)d_out;

    char* ws = (char*)d_ws;
    size_t off = 0;
    float*    bufA  = (float*)(ws + off); off += (size_t)N * GAT_D * 4;  // layer io / acc
    float*    hbuf  = (float*)(ws + off); off += (size_t)N * GAT_D * 4;  // h = x@W
    float*    ebuf  = (float*)(ws + off); off += (size_t)Etot * 4;       // logit -> exp (in place)
    unsigned* mEnc  = (unsigned*)(ws + off); off += (size_t)N * 4;       // encoded seg-max
    float*    sSum  = (float*)(ws + off); off += (size_t)N * 4;          // seg-sum (adjacent to mEnc)
    float*    asBuf = (float*)(ws + off); off += (size_t)N * 4;
    float*    adBuf = (float*)(ws + off); off += (size_t)N * 4;

    const int gGemm  = (N + 127) / 128;
    const int gAlpha = (N + 7) / 8;
    const int gEdge  = (Etot + 255) / 256;
    const int gAggr  = (int)(((long)Etot * 32 + 255) / 256);
    const int gNode2 = (2 * N + 255) / 256;
    const int gND    = (N * GAT_D + 255) / 256;

    for (int sIdx = 0; sIdx < 4; ++sIdx) {
        const int p = (sIdx < 2) ? 0 : 1;   // 0,1 -> pos params; 2,3 -> neg params
        const int* esrc = edges[sIdx];
        const int* edst = edges[sIdx] + E;
        for (int l = 0; l < GAT_L; ++l) {
            const float* Wl  = Wset[p]  + (size_t)l * GAT_D * GAT_D;
            const float* asl = ASset[p] + (size_t)l * GAT_D;
            const float* adl = ADset[p] + (size_t)l * GAT_D;
            const float* bl  = Bset[p]  + (size_t)l * GAT_D;
            const float* cur = (l == 0) ? X : bufA;
            float* nxt = (l == GAT_L - 1) ? (out + (size_t)sIdx * N * GAT_D) : bufA;

            gat_gemm<<<gGemm, 256, 0, stream>>>(cur, Wl, hbuf, N);
            node_alpha<<<gAlpha, 256, 0, stream>>>(hbuf, asl, adl, asBuf, adBuf, N);
            zero_u32<<<gNode2, 256, 0, stream>>>(mEnc, (long)2 * N); // mEnc + sSum adjacent
            edge_logits<<<gEdge, 256, 0, stream>>>(esrc, edst, asBuf, adBuf, ebuf, mEnc, Etot, E);
            edge_exp<<<gEdge, 256, 0, stream>>>(edst, ebuf, mEnc, sSum, Etot, E);
            zero_u32<<<gND, 256, 0, stream>>>((unsigned*)bufA, (long)N * GAT_D); // acc=bufA (cur dead after gemm)
            edge_aggregate<<<gAggr, 256, 0, stream>>>(esrc, edst, hbuf, ebuf, sSum, bufA, Etot, E);
            gat_finalize<<<gND, 256, 0, stream>>>(bufA, bl, pw, nxt, N * GAT_D);
        }
    }
}
